// Generator_77764677862116
// MI455X (gfx1250) — matrix operand order repaired and checked by reading, not compiled, not run on hardware
//
#include <hip/hip_runtime.h>
#include <hip/hip_bf16.h>

// ---------------------------------------------------------------------------
// Types for WMMA
// ---------------------------------------------------------------------------
typedef __attribute__((ext_vector_type(16))) __bf16       v16bf;
typedef __attribute__((ext_vector_type(8)))  float        v8f;
typedef __attribute__((ext_vector_type(4))) unsigned int  uint4v;
typedef __attribute__((__vector_size__(4 * sizeof(int)))) int gv4i;  // matches async-LDS builtin param

union ABFrag { v16bf bf; uint4v u[2]; };

#define L_LAYERS 4
#define DIM      1024
#define HEADS    4
#define DH       256
#define MLP_H    4096
#define NSEQ     4608          // 8*24*24 tokens
#define BATCH    2
#define MTOK     (BATCH * NSEQ)   // 9216
#define SCALE_QK 0.03125f         // 1/sqrt(1024)

#define BM 128
#define BN 128
#define BK 32
#define LDSS 48   // padded LDS row stride (elems), keeps 16B alignment

#define AS1 __attribute__((address_space(1)))
#define AS3 __attribute__((address_space(3)))

// gfx1250 async global->LDS path (guarded: falls back to sync staging if the
// toolchain doesn't expose the builtins)
#if defined(__AMDGCN__) && __has_builtin(__builtin_amdgcn_global_load_async_to_lds_b128)
#define USE_ASYNC_LDS 1
#else
#define USE_ASYNC_LDS 0
#endif

#if defined(__AMDGCN__) && __has_builtin(__builtin_amdgcn_s_wait_asynccnt)
#define WAIT_ASYNC(n) __builtin_amdgcn_s_wait_asynccnt(n)
#else
#define WAIT_ASYNC(n) asm volatile("s_wait_asynccnt %0" ::"i"(n) : "memory")
#endif

// ---------------------------------------------------------------------------
// Generic bf16 WMMA GEMM:  C(MxN) = alpha * A(MxK) * Bt(NxK)^T  [+bias][+resid]
//   epi 0: Cf = v                      (f32 out)
//   epi 1: Cb = bf16(v + bias?)        (bf16 out)
//   epi 2: Cf = resid + v + bias       (f32 out, residual add)
//   epi 3: Cb = bf16(gelu(v + bias))   (bf16 out, exact GELU)
// M,N multiples of 128; K multiple of 32; lda/ldb multiples of 8.
// ---------------------------------------------------------------------------
__global__ __launch_bounds__(256)
void gemm_bf16_kernel(const __bf16* __restrict__ A, int lda,
                      const __bf16* __restrict__ Bt, int ldb,
                      float* __restrict__ Cf, __bf16* __restrict__ Cb, int ldc,
                      const float* __restrict__ bias,
                      const float* __restrict__ resid, int ldr,
                      int K, float alpha, int epi)
{
    __shared__ __bf16 sA[2][BM * LDSS];
    __shared__ __bf16 sB[2][BN * LDSS];

    const int tid   = threadIdx.x;
    const int lane  = tid & 31;
    const int wave  = tid >> 5;
    const int waveM = wave >> 2;          // 0..1   (64-row slab)
    const int waveN = wave & 3;           // 0..3   (32-col slab)
    const int m0    = blockIdx.y * BM;
    const int n0    = blockIdx.x * BN;

    v8f acc[4][2];
#pragma unroll
    for (int mt = 0; mt < 4; ++mt)
#pragma unroll
        for (int nt = 0; nt < 2; ++nt)
#pragma unroll
            for (int e = 0; e < 8; ++e) acc[mt][nt][e] = 0.0f;

    const int ldRow  = tid >> 1;          // 0..127
    const int ldHalf = (tid & 1) * 16;    // 0 or 16 elems
    const __bf16* gAbase = A  + (size_t)(m0 + ldRow) * lda + ldHalf;
    const __bf16* gBbase = Bt + (size_t)(n0 + ldRow) * ldb + ldHalf;

    // fragment compute on one LDS buffer (16-bit A 16x32 lane layout)
    auto compute_tile = [&](int buf) {
        ABFrag af[4], bfr[2];
#pragma unroll
        for (int mt = 0; mt < 4; ++mt) {
            int row = waveM * 64 + mt * 16 + (lane & 15);
            int c   = (lane >> 4) * 8;
            af[mt].u[0] = *reinterpret_cast<const uint4v*>(&sA[buf][row * LDSS + c]);
            af[mt].u[1] = *reinterpret_cast<const uint4v*>(&sA[buf][row * LDSS + c + 16]);
        }
#pragma unroll
        for (int nt = 0; nt < 2; ++nt) {
            int row = waveN * 32 + nt * 16 + (lane & 15);
            int c   = (lane >> 4) * 8;
            bfr[nt].u[0] = *reinterpret_cast<const uint4v*>(&sB[buf][row * LDSS + c]);
            bfr[nt].u[1] = *reinterpret_cast<const uint4v*>(&sB[buf][row * LDSS + c + 16]);
        }
#pragma unroll
        for (int mt = 0; mt < 4; ++mt)
#pragma unroll
            for (int nt = 0; nt < 2; ++nt)
                acc[mt][nt] = __builtin_amdgcn_wmma_f32_16x16x32_bf16(
                    false, af[mt].bf, false, bfr[nt].bf,
                    (short)0, acc[mt][nt], false, false);
    };

    const int nk = K / BK;

#if USE_ASYNC_LDS
    // async global->LDS (ASYNCcnt) with double buffering:
    // issue tile i+1 into buf^1, wait oldest 4 (tile i), barrier, compute.
    auto issue_tile = [&](int buf, int k0) {
        AS1 gv4i* ga = (AS1 gv4i*)(uintptr_t)(gAbase + k0);
        AS1 gv4i* gb = (AS1 gv4i*)(uintptr_t)(gBbase + k0);
        AS3 gv4i* la0 = (AS3 gv4i*)(unsigned)(uintptr_t)&sA[buf][ldRow * LDSS + ldHalf];
        AS3 gv4i* la1 = (AS3 gv4i*)(unsigned)(uintptr_t)&sA[buf][ldRow * LDSS + ldHalf + 8];
        AS3 gv4i* lb0 = (AS3 gv4i*)(unsigned)(uintptr_t)&sB[buf][ldRow * LDSS + ldHalf];
        AS3 gv4i* lb1 = (AS3 gv4i*)(unsigned)(uintptr_t)&sB[buf][ldRow * LDSS + ldHalf + 8];
        __builtin_amdgcn_global_load_async_to_lds_b128(ga,     la0, 0, 0);
        __builtin_amdgcn_global_load_async_to_lds_b128(ga + 1, la1, 0, 0);
        __builtin_amdgcn_global_load_async_to_lds_b128(gb,     lb0, 0, 0);
        __builtin_amdgcn_global_load_async_to_lds_b128(gb + 1, lb1, 0, 0);
    };

    issue_tile(0, 0);
    for (int i = 0; i < nk; ++i) {
        const int cur = i & 1;
        const int kn  = (i + 1 < nk) ? (i + 1) : i;   // clamp: harmless reload
        issue_tile(cur ^ 1, kn * BK);
        WAIT_ASYNC(4);        // our 4 oldest (current tile) have landed in LDS
        __syncthreads();      // all waves' current tile visible
        compute_tile(cur);
        __syncthreads();      // done reading buf 'cur' before it is rewritten
    }
#else
    // sync fallback: global -> regs -> LDS, single buffer
    for (int i = 0; i < nk; ++i) {
        const int k0 = i * BK;
        const uint4v* gA = reinterpret_cast<const uint4v*>(gAbase + k0);
        const uint4v* gB = reinterpret_cast<const uint4v*>(gBbase + k0);
        uint4v a0 = gA[0], a1 = gA[1];
        uint4v b0 = gB[0], b1 = gB[1];
        __syncthreads();
        *reinterpret_cast<uint4v*>(&sA[0][ldRow * LDSS + ldHalf])     = a0;
        *reinterpret_cast<uint4v*>(&sA[0][ldRow * LDSS + ldHalf + 8]) = a1;
        *reinterpret_cast<uint4v*>(&sB[0][ldRow * LDSS + ldHalf])     = b0;
        *reinterpret_cast<uint4v*>(&sB[0][ldRow * LDSS + ldHalf + 8]) = b1;
        __syncthreads();
        compute_tile(0);
    }
#endif

    // epilogue: lane l owns rows (l>>4)*8 + i, col l&15 within each 16x16 tile
#pragma unroll
    for (int mt = 0; mt < 4; ++mt) {
#pragma unroll
        for (int nt = 0; nt < 2; ++nt) {
            const int gn     = n0 + waveN * 32 + nt * 16 + (lane & 15);
            const int gmBase = m0 + waveM * 64 + mt * 16 + (lane >> 4) * 8;
            const float bv   = bias ? bias[gn] : 0.0f;
#pragma unroll
            for (int i = 0; i < 8; ++i) {
                const int gm = gmBase + i;
                float v = acc[mt][nt][i] * alpha + bv;
                if (epi == 0) {
                    Cf[(size_t)gm * ldc + gn] = v;
                } else if (epi == 2) {
                    Cf[(size_t)gm * ldc + gn] = resid[(size_t)gm * ldr + gn] + v;
                } else {
                    if (epi == 3)
                        v = 0.5f * v * (1.0f + erff(v * 0.70710678f));
                    Cb[(size_t)gm * ldc + gn] = (__bf16)v;
                }
            }
        }
    }
}

// ---------------------------------------------------------------------------
// Weight transpose + cast: Wt[n*K + k] = bf16(W[k*N + n])
// ---------------------------------------------------------------------------
__global__ __launch_bounds__(256)
void transpose_cast_kernel(const float* __restrict__ W, __bf16* __restrict__ Wt,
                           int K, int N)
{
    int n = blockIdx.x * 32 + (threadIdx.x & 31);
    int k = blockIdx.y * 8  + (threadIdx.x >> 5);
    Wt[(size_t)n * K + k] = (__bf16)W[(size_t)k * N + n];
}

// ---------------------------------------------------------------------------
// Patch embed (stride==kernel conv3d) + bias + pos  -> x (f32, [b, n, DIM])
// ---------------------------------------------------------------------------
__global__ __launch_bounds__(256)
void patch_embed_kernel(const float* __restrict__ img, const float* __restrict__ cw,
                        const float* __restrict__ cb, const float* __restrict__ pos,
                        float* __restrict__ x)
{
    const int c = blockIdx.x * 256 + threadIdx.x;   // output channel
    const int n = blockIdx.y;                        // token = d*576 + h*24 + w
    const int b = blockIdx.z;
    const int d = n / 576;
    const int hw = n % 576;
    const int h = hw / 24;
    const int w = hw % 24;

    const float* wp = cw + (size_t)c * 256;          // (2,2,8,8) flattened
    float acc = cb[c];
    int idx = 0;
#pragma unroll
    for (int i = 0; i < 2; ++i)
#pragma unroll
        for (int kd = 0; kd < 2; ++kd) {
            const float* base =
                img + ((((size_t)b * 2 + i) * 16 + (d * 2 + kd)) * 192 + h * 8) * 192 + w * 8;
            for (int kh = 0; kh < 8; ++kh) {
                const float* r = base + kh * 192;
#pragma unroll
                for (int kw = 0; kw < 8; ++kw) acc += r[kw] * wp[idx++];
            }
        }
    x[((size_t)b * NSEQ + n) * DIM + c] = acc + pos[(size_t)n * DIM + c];
}

// ---------------------------------------------------------------------------
// LayerNorm over last dim (1024).  Writes bf16 (outb) or f32 (outf).
// ---------------------------------------------------------------------------
__global__ __launch_bounds__(256)
void layernorm_kernel(const float* __restrict__ x, const float* __restrict__ g,
                      const float* __restrict__ bta,
                      __bf16* __restrict__ outb, float* __restrict__ outf)
{
    __shared__ float red[256];
    const int row = blockIdx.x;
    const int t   = threadIdx.x;
    const float* xr = x + (size_t)row * DIM;

    float v[4];
    float s = 0.0f;
#pragma unroll
    for (int i = 0; i < 4; ++i) { v[i] = xr[t * 4 + i]; s += v[i]; }
    red[t] = s; __syncthreads();
    for (int o = 128; o > 0; o >>= 1) { if (t < o) red[t] += red[t + o]; __syncthreads(); }
    const float mu = red[0] * (1.0f / DIM);
    __syncthreads();

    float s2 = 0.0f;
#pragma unroll
    for (int i = 0; i < 4; ++i) { float d0 = v[i] - mu; s2 += d0 * d0; }
    red[t] = s2; __syncthreads();
    for (int o = 128; o > 0; o >>= 1) { if (t < o) red[t] += red[t + o]; __syncthreads(); }
    const float rstd = rsqrtf(red[0] * (1.0f / DIM) + 1e-5f);

#pragma unroll
    for (int i = 0; i < 4; ++i) {
        const int c = t * 4 + i;
        const float y = (v[i] - mu) * rstd * g[c] + bta[c];
        if (outb) outb[(size_t)row * DIM + c] = (__bf16)y;
        else      outf[(size_t)row * DIM + c] = y;
    }
}

// ---------------------------------------------------------------------------
// Split QKV (bf16 [b,n,3072]) -> Q,K [bh,n,dh] and V transposed [bh,dh,n]
// ---------------------------------------------------------------------------
__global__ __launch_bounds__(256)
void split_qkv_kernel(const __bf16* __restrict__ QKV,
                      __bf16* __restrict__ Q, __bf16* __restrict__ Kb,
                      __bf16* __restrict__ Vt)
{
    const int c = blockIdx.x * 256 + threadIdx.x;  // 0..3071
    const int n = blockIdx.y;
    const int b = blockIdx.z;
    const __bf16 val = QKV[((size_t)b * NSEQ + n) * (3 * DIM) + c];
    const int j  = c >> 10;          // 0=q 1=k 2=v
    const int hd = c & 1023;
    const int h  = hd >> 8;
    const int d  = hd & 255;
    const size_t bh = (size_t)(b * HEADS + h);
    if (j == 0)      Q [(bh * NSEQ + n) * DH + d] = val;
    else if (j == 1) Kb[(bh * NSEQ + n) * DH + d] = val;
    else             Vt[bh * (size_t)DH * NSEQ + (size_t)d * NSEQ + n] = val;
}

// ---------------------------------------------------------------------------
// Row softmax over n=4608: f32 scores -> bf16 probs
// ---------------------------------------------------------------------------
__global__ __launch_bounds__(256)
void softmax_kernel(const float* __restrict__ S, __bf16* __restrict__ P)
{
    __shared__ float red[256];
    const int row = blockIdx.x;
    const int t   = threadIdx.x;
    const float* sr = S + (size_t)row * NSEQ;

    float v[18];
    float mx = -3.4e38f;
#pragma unroll
    for (int i = 0; i < 18; ++i) { v[i] = sr[i * 256 + t]; mx = fmaxf(mx, v[i]); }
    red[t] = mx; __syncthreads();
    for (int o = 128; o > 0; o >>= 1) { if (t < o) red[t] = fmaxf(red[t], red[t + o]); __syncthreads(); }
    mx = red[0]; __syncthreads();

    float sum = 0.0f;
#pragma unroll
    for (int i = 0; i < 18; ++i) { v[i] = expf(v[i] - mx); sum += v[i]; }
    red[t] = sum; __syncthreads();
    for (int o = 128; o > 0; o >>= 1) { if (t < o) red[t] += red[t + o]; __syncthreads(); }
    const float inv = 1.0f / red[0];
#pragma unroll
    for (int i = 0; i < 18; ++i)
        P[(size_t)row * NSEQ + i * 256 + t] = (__bf16)(v[i] * inv);
}

// ---------------------------------------------------------------------------
// Host side
// ---------------------------------------------------------------------------
extern "C" void kernel_launch(void* const* d_in, const int* in_sizes, int n_in,
                              void* d_out, int out_size, void* d_ws, size_t ws_size,
                              hipStream_t stream)
{
    const float* img    = (const float*)d_in[0];
    const float* conv_w = (const float*)d_in[1];
    const float* conv_b = (const float*)d_in[2];
    const float* pos    = (const float*)d_in[3];
    const float* ln1_g  = (const float*)d_in[4];
    const float* ln1_b  = (const float*)d_in[5];
    const float* qkv_w  = (const float*)d_in[6];
    const float* out_w  = (const float*)d_in[7];
    const float* out_b  = (const float*)d_in[8];
    const float* ln2_g  = (const float*)d_in[9];
    const float* ln2_b  = (const float*)d_in[10];
    const float* fc1_w  = (const float*)d_in[11];
    const float* fc1_b  = (const float*)d_in[12];
    const float* fc2_w  = (const float*)d_in[13];
    const float* fc2_b  = (const float*)d_in[14];
    const float* lnf_g  = (const float*)d_in[15];
    const float* lnf_b  = (const float*)d_in[16];

    char* ws = (char*)d_ws;
    size_t off = 0;
    auto alloc = [&](size_t bytes) -> char* {
        char* p = ws + off;
        off += (bytes + 255) & ~(size_t)255;
        return p;
    };

    __bf16* wt_qkv = (__bf16*)alloc((size_t)L_LAYERS * 3 * DIM * DIM * 2);
    __bf16* wt_out = (__bf16*)alloc((size_t)L_LAYERS * DIM * DIM * 2);
    __bf16* wt_fc1 = (__bf16*)alloc((size_t)L_LAYERS * MLP_H * DIM * 2);
    __bf16* wt_fc2 = (__bf16*)alloc((size_t)L_LAYERS * DIM * MLP_H * 2);
    float*  x      = (float* )alloc((size_t)MTOK * DIM * 4);
    __bf16* xb     = (__bf16*)alloc((size_t)MTOK * DIM * 2);
    __bf16* big    = (__bf16*)alloc((size_t)MTOK * MLP_H * 2);  // QKV / MLP hidden
    __bf16* Q      = (__bf16*)alloc((size_t)BATCH * HEADS * NSEQ * DH * 2);
    __bf16* Kb     = (__bf16*)alloc((size_t)BATCH * HEADS * NSEQ * DH * 2);
    __bf16* Vt     = (__bf16*)alloc((size_t)BATCH * HEADS * NSEQ * DH * 2);
    __bf16* O      = (__bf16*)alloc((size_t)MTOK * DIM * 2);
    float*  Sbuf   = (float* )alloc((size_t)NSEQ * NSEQ * 4);
    __bf16* Pbuf   = (__bf16*)alloc((size_t)NSEQ * NSEQ * 2);

    auto gemm = [&](const __bf16* A, int lda, const __bf16* Bt, int ldb,
                    float* Cf, __bf16* Cb, int ldc,
                    const float* bias, const float* resid, int ldr,
                    int M, int N, int K, float alpha, int epi) {
        dim3 grid(N / BN, M / BM);
        gemm_bf16_kernel<<<grid, 256, 0, stream>>>(A, lda, Bt, ldb, Cf, Cb, ldc,
                                                   bias, resid, ldr, K, alpha, epi);
    };
    auto transpose = [&](const float* W, __bf16* Wt, int K, int N) {
        transpose_cast_kernel<<<dim3(N / 32, K / 8), 256, 0, stream>>>(W, Wt, K, N);
    };

    // ---- weight prep (f32 KxN -> bf16 NxK) ----
    for (int l = 0; l < L_LAYERS; ++l) {
        transpose(qkv_w + (size_t)l * DIM * 3 * DIM, wt_qkv + (size_t)l * 3 * DIM * DIM, DIM, 3 * DIM);
        transpose(out_w + (size_t)l * DIM * DIM,     wt_out + (size_t)l * DIM * DIM,     DIM, DIM);
        transpose(fc1_w + (size_t)l * DIM * MLP_H,   wt_fc1 + (size_t)l * MLP_H * DIM,   DIM, MLP_H);
        transpose(fc2_w + (size_t)l * MLP_H * DIM,   wt_fc2 + (size_t)l * DIM * MLP_H,   MLP_H, DIM);
    }

    // ---- patch embed + pos ----
    patch_embed_kernel<<<dim3(DIM / 256, NSEQ, BATCH), 256, 0, stream>>>(
        img, conv_w, conv_b, pos, x);

    // ---- transformer layers ----
    for (int l = 0; l < L_LAYERS; ++l) {
        // LN1 -> bf16
        layernorm_kernel<<<MTOK, 256, 0, stream>>>(x, ln1_g + l * DIM, ln1_b + l * DIM, xb, nullptr);

        // QKV = x1 @ qkv_w   (bf16 out, no bias)
        gemm(xb, DIM, wt_qkv + (size_t)l * 3 * DIM * DIM, DIM,
             nullptr, big, 3 * DIM, nullptr, nullptr, 0,
             MTOK, 3 * DIM, DIM, 1.0f, 1);

        split_qkv_kernel<<<dim3(12, NSEQ, BATCH), 256, 0, stream>>>(big, Q, Kb, Vt);

        // attention per (b,h), reusing S/P scratch via stream ordering
        for (int bh = 0; bh < BATCH * HEADS; ++bh) {
            const __bf16* Qh  = Q  + (size_t)bh * NSEQ * DH;
            const __bf16* Kh  = Kb + (size_t)bh * NSEQ * DH;
            const __bf16* Vth = Vt + (size_t)bh * DH * NSEQ;
            const int b = bh / HEADS, h = bh % HEADS;

            // S = scale * Q @ K^T
            gemm(Qh, DH, Kh, DH, Sbuf, nullptr, NSEQ, nullptr, nullptr, 0,
                 NSEQ, NSEQ, DH, SCALE_QK, 0);
            softmax_kernel<<<NSEQ, 256, 0, stream>>>(Sbuf, Pbuf);
            // O[b, :, h*DH:(h+1)*DH] = P @ V   (merged-head layout via ldc)
            gemm(Pbuf, NSEQ, Vth, NSEQ,
                 nullptr, O + (size_t)b * NSEQ * DIM + h * DH, DIM,
                 nullptr, nullptr, 0, NSEQ, DH, NSEQ, 1.0f, 1);
        }

        // x = x + O @ out_w + out_b   (in-place f32)
        gemm(O, DIM, wt_out + (size_t)l * DIM * DIM, DIM,
             x, nullptr, DIM, out_b + l * DIM, x, DIM,
             MTOK, DIM, DIM, 1.0f, 2);

        // LN2 -> bf16
        layernorm_kernel<<<MTOK, 256, 0, stream>>>(x, ln2_g + l * DIM, ln2_b + l * DIM, xb, nullptr);

        // h = gelu(x2 @ fc1_w + fc1_b)  (bf16 out)
        gemm(xb, DIM, wt_fc1 + (size_t)l * MLP_H * DIM, DIM,
             nullptr, big, MLP_H, fc1_b + l * MLP_H, nullptr, 0,
             MTOK, MLP_H, DIM, 1.0f, 3);

        // x = x + h @ fc2_w + fc2_b   (in-place f32)
        gemm(big, MLP_H, wt_fc2 + (size_t)l * DIM * MLP_H, MLP_H,
             x, nullptr, DIM, fc2_b + l * DIM, x, DIM,
             MTOK, DIM, MLP_H, 1.0f, 2);
    }

    // ---- final LN -> f32 output ----
    layernorm_kernel<<<MTOK, 256, 0, stream>>>(x, lnf_g, lnf_b, nullptr, (float*)d_out);
}
